// HMM_7559142441658
// MI455X (gfx1250) — matrix operand order repaired and checked by reading, not compiled, not run on hardware
//
#include <hip/hip_runtime.h>
#include <hip/hip_bf16.h>

typedef __attribute__((ext_vector_type(16))) _Float16 v16h;
typedef __attribute__((ext_vector_type(8)))  float    v8f;

#define NSTATE 128
#define SEQLEN 4096
#define BATCH  64
#define ROWS   16            // batch rows per workgroup (= WMMA M)
#define NBLK   (BATCH / ROWS) // 4 workgroups per direction

// ---------------- workspace layout ----------------
// alpha  : SEQLEN * BATCH * NSTATE f32, tiled as [t][bc][row][n]  (134 MB)
// expT   : exp(T[j][i] - colmax[i])  f16  [j][i]                   (32 KB)
// expTtr : exp(T[i][k] - rowmax[i])  f16  [k][i]                   (32 KB)
// colmax : f32[128]   rowmax : f32[128]
#define ALPHA_ELEMS ((size_t)SEQLEN * BATCH * NSTATE)

// ============================================================
// Prep: build exp-shifted transition tables (constant across the scan)
// ============================================================
__global__ void hmm_prep_kernel(const float* __restrict__ Tm,
                                _Float16* __restrict__ expT,
                                _Float16* __restrict__ expTtr,
                                float* __restrict__ colmax,
                                float* __restrict__ rowmax) {
    int i = threadIdx.x;
    if (i >= NSTATE) return;
    // column max over j of T[j][i]
    float cm = -1e30f;
    for (int j = 0; j < NSTATE; ++j) cm = fmaxf(cm, Tm[j * NSTATE + i]);
    colmax[i] = cm;
    for (int j = 0; j < NSTATE; ++j)
        expT[j * NSTATE + i] = (_Float16)__expf(Tm[j * NSTATE + i] - cm);
    // row max over k of T[i][k]  (column max of T^T)
    float rm = -1e30f;
    for (int k = 0; k < NSTATE; ++k) rm = fmaxf(rm, Tm[i * NSTATE + k]);
    rowmax[i] = rm;
    for (int k = 0; k < NSTATE; ++k)
        expTtr[k * NSTATE + i] = (_Float16)__expf(Tm[i * NSTATE + k] - rm);
}

// ------------------------------------------------------------
// A-fragment (16x32 f16) load from LDS per ISA VGPR layout:
// lanes 0-15 row M=lane, K = kb..kb+7 and kb+16..kb+23 (kb = k0 or k0+8)
// ------------------------------------------------------------
__device__ __forceinline__ v16h load_a_frag(const _Float16* pS, int lane, int k0) {
    int ra = lane & 15;
    int kb = k0 + ((lane >> 4) << 3);
    v16h af;
#pragma unroll
    for (int i = 0; i < 8; ++i) {
        af[i]     = pS[ra * NSTATE + kb + i];
        af[8 + i] = pS[ra * NSTATE + kb + 16 + i];
    }
    return af;
}

// ============================================================
// Forward: alpha recursion; 4 blocks x 256 threads (8 waves)
// wave w owns output columns [16w, 16w+16)
// ============================================================
__global__ void __launch_bounds__(256)
hmm_forward_kernel(const int* __restrict__ obs,
                   const float* __restrict__ lip,   // log_initial_probs
                   const float* __restrict__ lem,   // log_emission_probs
                   const _Float16* __restrict__ expT,
                   const float* __restrict__ colmax,
                   float* __restrict__ alpha_g) {
    __shared__ __align__(16) float    aS[ROWS * NSTATE];  // alpha_t (f32)
    __shared__ __align__(16) _Float16 pS[ROWS * NSTATE];  // exp(alpha - m) (f16)
    __shared__ float mS[ROWS];
    __shared__ float eS[ROWS];

    const int tid   = threadIdx.x;
    const int lane  = tid & 31;
    const int wv    = tid >> 5;
    const int bc    = blockIdx.x;
    const int bbase = bc * ROWS;
    const int n0    = wv * 16;
    const int col   = lane & 15;
    const int hi    = lane >> 4;

    // ---- preload B fragments for this wave's column tile (kept in VGPRs
    //      across all 4096 steps) + per-lane column max ----
    v16h bf[4];
#pragma unroll
    for (int c = 0; c < 4; ++c)
#pragma unroll
        for (int i = 0; i < 16; ++i)
            bf[c][i] = expT[(c * 32 + hi * 8 + (i & 7) + ((i >> 3) << 4)) * NSTATE + (n0 + col)];
    const float cmLane = colmax[n0 + col];

    // ---- t = 0 init: alpha0 = lip[n] + e[b,0] ----
    {
        int idx8 = tid * 8;
        int row  = tid >> 4;
        float e0 = lem[obs[(size_t)(bbase + row) * SEQLEN]];
#pragma unroll
        for (int k = 0; k < 8; ++k) {
            int idx = idx8 + k;
            aS[idx] = lip[idx & (NSTATE - 1)] + e0;
        }
    }
    __syncthreads();

    for (int t = 1; t < SEQLEN; ++t) {
        // ---- phase 1: waves 0-3 row-max; waves 4-7 flush alpha[t-1] + eS ----
        if (tid < 128) {
            int row = tid >> 3, seg = tid & 7;
            const float* ap = aS + row * NSTATE + seg * 16;
            float v = ap[0];
#pragma unroll
            for (int c = 1; c < 16; ++c) v = fmaxf(v, ap[c]);
            v = fmaxf(v, __shfl_xor(v, 1));
            v = fmaxf(v, __shfl_xor(v, 2));
            v = fmaxf(v, __shfl_xor(v, 4));
            if (seg == 0) mS[row] = v;
        } else {
            int ctid = tid - 128;
            float4*       dst = (float4*)(alpha_g + ((size_t)(t - 1) * NBLK + bc) * (ROWS * NSTATE));
            const float4* src = (const float4*)aS;
#pragma unroll
            for (int k = 0; k < 4; ++k) dst[ctid + 128 * k] = src[ctid + 128 * k];
            if (ctid < ROWS)
                eS[ctid] = lem[obs[(size_t)(bbase + ctid) * SEQLEN + t]];
        }
        __syncthreads();

        // ---- phase 2: pS = exp(alpha - m) as f16 ----
        {
            int idx8 = tid * 8;
            int row  = tid >> 4;
            float m  = mS[row];
#pragma unroll
            for (int k = 0; k < 8; ++k)
                pS[idx8 + k] = (_Float16)__expf(aS[idx8 + k] - m);
        }
        __syncthreads();

        // ---- phase 3: WMMA GEMM + log + emission ----
        {
            v8f acc = {};
#pragma unroll
            for (int c = 0; c < 4; ++c) {
                v16h af = load_a_frag(pS, lane, c * 32);
                acc = __builtin_amdgcn_wmma_f32_16x16x32_f16(
                    false, af, false, bf[c], (short)0, acc, false, false);
            }
#pragma unroll
            for (int v = 0; v < 8; ++v) {
                int r = v + hi * 8;
                aS[r * NSTATE + n0 + col] = mS[r] + cmLane + __logf(acc[v]) + eS[r];
            }
        }
        __syncthreads();
    }

    // ---- flush alpha[T-1] ----
    {
        float4*       dst = (float4*)(alpha_g + ((size_t)(SEQLEN - 1) * NBLK + bc) * (ROWS * NSTATE));
        const float4* src = (const float4*)aS;
        for (int k = tid; k < 512; k += 256) dst[k] = src[k];
    }
}

// ============================================================
// Backward + fused gamma: beta recursion with expT^T, then
// gamma[t] = normalize(alpha[t] + beta[t]) written straight to out.
// ============================================================
__global__ void __launch_bounds__(256)
hmm_backward_kernel(const int* __restrict__ obs,
                    const float* __restrict__ lem,
                    const _Float16* __restrict__ expTtr,
                    const float* __restrict__ rowmax,
                    const float* __restrict__ alpha_g,
                    float* __restrict__ out) {
    __shared__ __align__(16) float    bS[ROWS * NSTATE];  // beta_t (f32)
    __shared__ __align__(16) _Float16 pS[ROWS * NSTATE];
    __shared__ float mS[ROWS];
    __shared__ float eS[ROWS];

    const int tid   = threadIdx.x;
    const int lane  = tid & 31;
    const int wv    = tid >> 5;
    const int bc    = blockIdx.x;
    const int bbase = bc * ROWS;
    const int n0    = wv * 16;
    const int col   = lane & 15;
    const int hi    = lane >> 4;

    v16h bf[4];
#pragma unroll
    for (int c = 0; c < 4; ++c)
#pragma unroll
        for (int i = 0; i < 16; ++i)
            bf[c][i] = expTtr[(c * 32 + hi * 8 + (i & 7) + ((i >> 3) << 4)) * NSTATE + (n0 + col)];
    const float cmLane = rowmax[n0 + col];

    // beta[T-1] = 0
    {
        int idx8 = tid * 8;
#pragma unroll
        for (int k = 0; k < 8; ++k) bS[idx8 + k] = 0.0f;
    }
    __syncthreads();

    // gamma helper (all values in registers; 16-lane shuffle logsumexp)
    auto emit_gamma = [&](int t) {
        int row  = tid >> 4;
        int nblk = (tid & 15) * 8;
        const float* ag = alpha_g + ((size_t)t * NBLK + bc) * (ROWS * NSTATE) + row * NSTATE + nblk;
        float g[8];
#pragma unroll
        for (int k = 0; k < 8; ++k) g[k] = ag[k] + bS[row * NSTATE + nblk + k];
        float mx = g[0];
#pragma unroll
        for (int k = 1; k < 8; ++k) mx = fmaxf(mx, g[k]);
        mx = fmaxf(mx, __shfl_xor(mx, 1));
        mx = fmaxf(mx, __shfl_xor(mx, 2));
        mx = fmaxf(mx, __shfl_xor(mx, 4));
        mx = fmaxf(mx, __shfl_xor(mx, 8));
        float s = 0.0f;
#pragma unroll
        for (int k = 0; k < 8; ++k) s += __expf(g[k] - mx);
        s += __shfl_xor(s, 1);
        s += __shfl_xor(s, 2);
        s += __shfl_xor(s, 4);
        s += __shfl_xor(s, 8);
        float lse = mx + __logf(s);
        float* op = out + ((size_t)(bbase + row) * SEQLEN + t) * NSTATE + nblk;
#pragma unroll
        for (int k = 0; k < 8; ++k) op[k] = g[k] - lse;
    };

    emit_gamma(SEQLEN - 1);
    __syncthreads();

    for (int t = SEQLEN - 2; t >= 0; --t) {
        // ---- phase 1: row-max of beta_{t+1}; eS = e[:, t+1] ----
        if (tid < 128) {
            int row = tid >> 3, seg = tid & 7;
            const float* bp = bS + row * NSTATE + seg * 16;
            float v = bp[0];
#pragma unroll
            for (int c = 1; c < 16; ++c) v = fmaxf(v, bp[c]);
            v = fmaxf(v, __shfl_xor(v, 1));
            v = fmaxf(v, __shfl_xor(v, 2));
            v = fmaxf(v, __shfl_xor(v, 4));
            if (seg == 0) mS[row] = v;
        } else if (tid < 128 + ROWS) {
            int r = tid - 128;
            eS[r] = lem[obs[(size_t)(bbase + r) * SEQLEN + (t + 1)]];
        }
        __syncthreads();

        // ---- phase 2: pS = exp(beta - m) ----
        {
            int idx8 = tid * 8;
            int row  = tid >> 4;
            float m  = mS[row];
#pragma unroll
            for (int k = 0; k < 8; ++k)
                pS[idx8 + k] = (_Float16)__expf(bS[idx8 + k] - m);
        }
        __syncthreads();

        // ---- phase 3: WMMA with expT^T; beta_t = m + e_{t+1} + rowmax + log(acc)
        {
            v8f acc = {};
#pragma unroll
            for (int c = 0; c < 4; ++c) {
                v16h af = load_a_frag(pS, lane, c * 32);
                acc = __builtin_amdgcn_wmma_f32_16x16x32_f16(
                    false, af, false, bf[c], (short)0, acc, false, false);
            }
#pragma unroll
            for (int v = 0; v < 8; ++v) {
                int r = v + hi * 8;
                bS[r * NSTATE + n0 + col] = mS[r] + eS[r] + cmLane + __logf(acc[v]);
            }
        }
        __syncthreads();

        // ---- phase 4: gamma[t] = normalize(alpha[t] + beta[t]) -> out ----
        emit_gamma(t);
        __syncthreads();
    }
}

// ============================================================
extern "C" void kernel_launch(void* const* d_in, const int* in_sizes, int n_in,
                              void* d_out, int out_size, void* d_ws, size_t ws_size,
                              hipStream_t stream) {
    const int*   obs = (const int*)d_in[0];
    const float* lip = (const float*)d_in[1];
    const float* Tm  = (const float*)d_in[2];
    const float* lem = (const float*)d_in[3];
    float*       out = (float*)d_out;

    float*    alpha_g = (float*)d_ws;
    _Float16* expT    = (_Float16*)((char*)d_ws + ALPHA_ELEMS * sizeof(float));
    _Float16* expTtr  = expT + NSTATE * NSTATE;
    float*    colmax  = (float*)(expTtr + NSTATE * NSTATE);
    float*    rowmax  = colmax + NSTATE;

    hipLaunchKernelGGL(hmm_prep_kernel, dim3(1), dim3(128), 0, stream,
                       Tm, expT, expTtr, colmax, rowmax);
    hipLaunchKernelGGL(hmm_forward_kernel, dim3(NBLK), dim3(256), 0, stream,
                       obs, lip, lem, expT, colmax, alpha_g);
    hipLaunchKernelGGL(hmm_backward_kernel, dim3(NBLK), dim3(256), 0, stream,
                       obs, lem, expTtr, rowmax, alpha_g, out);
}
